// RLSE_85341000172024
// MI455X (gfx1250) — matrix operand order repaired and checked by reading, not compiled, not run on hardware
//
#include <hip/hip_runtime.h>
#include <hip/hip_bf16.h>

// RLSE scan: one persistent workgroup keeps S (256x256 f32 = 256KB) in the
// 320KB WGP LDS for all 8192 sequential steps. x/f coefficient rows are
// double-buffered in LDS via GLOBAL_LOAD_ASYNC_TO_LDS_B128 (ASYNCcnt),
// hiding HBM latency behind the LDS-bound S update. Final projection
// x @ theta uses v_wmma_f32_16x16x32_f16 (8 WMMAs per 16-row tile).

#define RLSE_N       256
#define RLSE_T       8192
#define INV_GAMMA    1.0e-3f

typedef __attribute__((ext_vector_type(16))) _Float16 v16h;
typedef __attribute__((ext_vector_type(8)))  float    v8f;

// ---------------------------------------------------------------------------
// Kernel 1: sequential RLSE scan, single workgroup, S resident in LDS.
// LDS layout (floats):
//   S      [0      , 65536)   256x256 state matrix
//   theta  [65536  , 65792)
//   a      [65792  , 66048)   current coefficient row (x*f)
//   dsnap  [66048  , 66304)   diag(S) snapshot (old S)
//   ci     [66304  , 66560)   per-row coefficient d_i*s2/(1+d_i*s2)
//   wv     [66560  , 66816)   w = S @ a
//   red    [66816  , 66880)   cross-wave reduction partials
//   scal   [66880  , 66888)   [0]=s2, [1]=e
//   xbuf   [66888  , 67400)   double-buffered x row (2 x 256)
//   fbuf   [67400  , 67912)   double-buffered f row (2 x 256)
// Total 67912 floats = 271648 B < 320KB WGP LDS.
// ---------------------------------------------------------------------------
__global__ void __launch_bounds__(1024, 1)
rlse_scan_kernel(const float* __restrict__ x, const float* __restrict__ f,
                 const float* __restrict__ y, float* __restrict__ theta_out)
{
    extern __shared__ float lds[];
    float* S     = lds;
    float* theta = lds + 65536;
    float* a     = theta + RLSE_N;
    float* dsnap = a + RLSE_N;
    float* ci    = dsnap + RLSE_N;
    float* wv    = ci + RLSE_N;
    float* red   = wv + RLSE_N;      // 64 floats
    float* scal  = red + 64;         // 8 floats
    float* xbuf  = scal + 8;         // 2 * 256
    float* fbuf  = xbuf + 2 * RLSE_N;// 2 * 256

    const int tid  = (int)threadIdx.x;
    const int lane = tid & 31;
    const int wave = tid >> 5;       // 0..31
    const int i0   = tid >> 8;       // 0..3   (S-update row group)
    const int kcol = tid & 255;      //        (S-update fixed column)

    // S = I, theta = 0
    for (int idx = tid; idx < RLSE_N * RLSE_N; idx += 1024)
        S[idx] = ((idx >> 8) == (idx & 255)) ? 1.0f : 0.0f;
    if (tid < RLSE_N) theta[tid] = 0.0f;

    // Async stage of one x/f row into LDS buffer nb.
    // Waves 0,1 copy x; waves 2,3 copy f; one b128 per lane (16B), one
    // async instruction per issuing wave -> per-wave ASYNCcnt += 1.
    auto issue_stage = [&](int trow, int nb) {
        if (wave < 4) {
            const float* src = (wave < 2) ? (x + (size_t)trow * RLSE_N)
                                          : (f + (size_t)trow * RLSE_N);
            float* dstb = ((wave < 2) ? xbuf : fbuf) + nb * RLSE_N;
            const int elt = (((wave & 1) * 32) + lane) * 4;   // 4 floats/lane
            unsigned ldsoff =
                (unsigned)((const char*)(dstb + elt) - (const char*)lds);
            const float* gaddr = src + elt;
            asm volatile("global_load_async_to_lds_b128 %0, %1, off"
                         :: "v"(ldsoff), "v"(gaddr) : "memory");
        }
    };

    issue_stage(0, 0);               // prologue: row 0 -> buffer 0

    for (int t = 0; t < RLSE_T; ++t) {
        const int cur = t & 1;
        // Issue next row into the other buffer (dummy reload of row 0 on the
        // last step keeps ASYNCcnt discipline uniform for the constant wait).
        const int nxt = (t + 1 < RLSE_T) ? (t + 1) : 0;
        issue_stage(nxt, cur ^ 1);

        // Warm L2 two steps ahead for the async engine (global_prefetch_b8).
        if (t + 2 < RLSE_T) {
            if (wave == 4) __builtin_prefetch(x + (size_t)(t + 2) * RLSE_N + lane * 8, 0, 1);
            if (wave == 5) __builtin_prefetch(f + (size_t)(t + 2) * RLSE_N + lane * 8, 0, 1);
        }

        // <=1 outstanding per issuing wave => current buffer's copy complete;
        // the next-buffer copy stays in flight. Non-issuing waves: no-op.
        asm volatile("s_wait_asynccnt 0x1" ::: "memory");
        __syncthreads();

        const float* xb = xbuf + cur * RLSE_N;
        const float* fb = fbuf + cur * RLSE_N;

        // Phase A: a = x*f, snapshot diag(S), reduce s2 = a.a and a.theta
        float pa2 = 0.0f, pat = 0.0f;
        if (tid < RLSE_N) {
            float av = xb[tid] * fb[tid];
            a[tid]     = av;
            dsnap[tid] = S[tid * (RLSE_N + 1)];   // old diagonal
            pa2 = av * av;
            pat = av * theta[tid];                // uses OLD theta
        }
        #pragma unroll
        for (int off = 16; off > 0; off >>= 1) {
            pa2 += __shfl_xor(pa2, off, 32);
            pat += __shfl_xor(pat, off, 32);
        }
        if (wave < 8 && lane == 0) { red[wave] = pa2; red[32 + wave] = pat; }
        __syncthreads();
        if (tid == 0) {
            float s2s = 0.0f, adot = 0.0f;
            #pragma unroll
            for (int q = 0; q < 8; ++q) { s2s += red[q]; adot += red[32 + q]; }
            scal[0] = s2s;
            scal[1] = y[t] - adot;                // e = b - a.theta_old
        }
        __syncthreads();
        const float s2 = scal[0];
        const float e  = scal[1];

        // Phase B: w = S @ a (old S). Wave w handles rows [8w, 8w+8);
        // lanes stride columns -> conflict-free LDS reads.
        {
            const int rbase = wave * 8;
            #pragma unroll
            for (int rr = 0; rr < 8; ++rr) {
                const float* Srow = S + (rbase + rr) * RLSE_N;
                float p = 0.0f;
                #pragma unroll
                for (int j = 0; j < 8; ++j) {
                    int c = lane + j * 32;
                    p = fmaf(Srow[c], a[c], p);
                }
                #pragma unroll
                for (int off = 16; off > 0; off >>= 1) p += __shfl_xor(p, off, 32);
                if (lane == 0) wv[rbase + rr] = p;
            }
        }
        __syncthreads();

        // Phase C: theta <- (theta + w*e)/gamma ; per-row S coefficient
        if (tid < RLSE_N) {
            theta[tid] = (theta[tid] + wv[tid] * e) * INV_GAMMA;
            float dn = dsnap[tid];
            ci[tid] = dn * s2 / fmaf(dn, s2, 1.0f);  // d_i*s2/x2_i
        }
        __syncthreads();

        // Phase D: S[i][k] = (S[i][k] - ci[i]*d[k]) / gamma
        // kcol fixed per thread (conflict-free), ci[i] broadcast per wave.
        {
            const float dk = dsnap[kcol];
            #pragma unroll 4
            for (int it = 0; it < 64; ++it) {
                int i   = i0 + it * 4;
                int idx = i * RLSE_N + kcol;
                S[idx] = fmaf(-ci[i], dk, S[idx]) * INV_GAMMA;
            }
        }
        __syncthreads();
    }

    // Drain the dummy in-flight async copy before ending the wave.
    asm volatile("s_wait_asynccnt 0x0" ::: "memory");
    if (tid < RLSE_N) theta_out[tid] = theta[tid];
}

// ---------------------------------------------------------------------------
// Kernel 2: out[m] = sum_n x[m][n] * theta[n]  via v_wmma_f32_16x16x32_f16.
// One wave per 16-row tile; theta replicated across B's 16 columns so every
// D column holds the matvec result. Fully unrolled: 8 WMMAs per tile.
// ---------------------------------------------------------------------------
__global__ void __launch_bounds__(256)
rlse_proj_kernel(const float* __restrict__ x, const float* __restrict__ theta,
                 float* __restrict__ out)
{
    const int lane  = (int)threadIdx.x & 31;
    const int gwave = (int)blockIdx.x * 8 + ((int)threadIdx.x >> 5);
    const int row_base = gwave * 16;          // 512 tiles over 8192 rows
    const int m  = lane & 15;                 // A-fragment row (M)
    const int hi = lane >> 4;                 // lane half selects K block

    v8f c = {};
    #pragma unroll
    for (int k0 = 0; k0 < RLSE_N; k0 += 32) {
        // A: 16x32 f16, lane layout per ISA 7.12.2:
        //   elem j<8  -> K = 8*hi + j ; elem j>=8 -> K = 8*hi + j + 8
        const float* arow = x + (size_t)(row_base + m) * RLSE_N + k0;
        const int abase = hi * 8;
        v16h afrag, bfrag;
        #pragma unroll
        for (int j = 0; j < 8; ++j) {
            afrag[j]     = (_Float16)arow[abase + j];
            afrag[j + 8] = (_Float16)arow[abase + 16 + j];
        }
        // B: 32x16 f16, lane L holds N=L%16, elem j<8 -> K = 8*hi + j ; elem j>=8 -> K = 8*hi + j + 8.
        // B[k][n] = theta[k0+k] for all n (column-replicated vector).
        const int bbase = hi * 8;
        #pragma unroll
        for (int j = 0; j < 8; ++j) {
            bfrag[j]     = (_Float16)theta[k0 + bbase + j];
            bfrag[j + 8] = (_Float16)theta[k0 + bbase + 16 + j];
        }

        c = __builtin_amdgcn_wmma_f32_16x16x32_f16(
                /*neg_a=*/false, afrag, /*neg_b=*/false, bfrag,
                /*c_mod=*/(short)0, c, /*reuse_a=*/false, /*reuse_b=*/false);
    }

    // D layout: VGPR r holds M=r (lanes 0-15) / M=r+8 (lanes 16-31), N=lane%16.
    // All N columns are identical; lanes with N==0 write the 16 row results.
    if ((lane & 15) == 0) {
        const int mb = row_base + hi * 8;
        #pragma unroll
        for (int r = 0; r < 8; ++r) out[mb + r] = c[r];
    }
}

// ---------------------------------------------------------------------------
extern "C" void kernel_launch(void* const* d_in, const int* in_sizes, int n_in,
                              void* d_out, int out_size, void* d_ws, size_t ws_size,
                              hipStream_t stream)
{
    (void)in_sizes; (void)n_in; (void)out_size; (void)ws_size;
    const float* x = (const float*)d_in[0];   // (4,2048,256) f32
    const float* f = (const float*)d_in[1];   // (4,2048,256) f32
    const float* y = (const float*)d_in[2];   // (4,2048,1)   f32
    float* out     = (float*)d_out;           // (4,2048,1)   f32
    float* theta_g = (float*)d_ws;            // 256 f32 scratch

    // 271648 bytes dynamic LDS: needs the >64KB dynamic-shared opt-in at
    // runtime (hipFuncAttributeMaxDynamicSharedMemorySize); fits gfx1250's
    // 320KB WGP LDS.
    const size_t lds_bytes = (size_t)(65536 + 5 * RLSE_N + 64 + 8 + 4 * RLSE_N)
                             * sizeof(float);
    rlse_scan_kernel<<<1, 1024, lds_bytes, stream>>>(x, f, y, theta_g);

    // 8192 rows / 16 per wave = 512 waves = 64 blocks x 8 waves.
    rlse_proj_kernel<<<64, 256, 0, stream>>>(x, theta_g, out);
}
